// InterEdgeAtten_4870492913914
// MI455X (gfx1250) — matrix operand order repaired and checked by reading, not compiled, not run on hardware
//
#include <hip/hip_runtime.h>
#include <hip/hip_bf16.h>

// ============================================================================
// InterEdgeAttention on MI455X (gfx1250, wave32, WMMA)
//
// Math: softmax(sim).sum(-1) == 1 identically (masked rows become uniform
// 1/256 which also sums to 1), so the reference reduces exactly to
//   out = ( (IE@Wv) * sigmoid(IE@Wg + bg)[gate-perm] ) @ Wo + bo
// with gate-perm gcol(c) = (c%64)*4 + c/64  (the "(d h)" vs "(h d)" split).
// q/k/Web/mask/softmax are dead. Remaining: 12.9 GFLOP, 67 MB HBM traffic
// (~2.9 us @ 23.3 TB/s). bf16 WMMA (16x16x32, f32 accum) puts math at the
// memory floor. MT=64 rows/block amortizes each weight B-fragment over 4
// M-tiles, halving L2 weight re-fetch (~400 MB -> ~200 MB across the grid).
// LDS: 16 KB X(bf16) + 32 KB gate(bf16) + 32 KB M(bf16) = 80 KB (< 320 KB/WGP).
// ============================================================================

typedef __attribute__((ext_vector_type(16))) __bf16       bf16x16;
typedef __attribute__((ext_vector_type(8)))  float        f32x8;
typedef __attribute__((ext_vector_type(4)))  unsigned int u32x4;
typedef __attribute__((ext_vector_type(4)))  float        f32x4;

#define EDIM   128   // edge dim (input & output channels)
#define INNERD 256   // H*DH
#define MT     64    // rows of IE per block
#define NMT    4     // M-tiles per block (MT/16)

union FragU { bf16x16 v; u32x4 q[2]; };

// A-matrix 16x32 bf16 per-lane layout (contiguous K, matching B): lane<16 holds
// K=k0..k0+15; lane>=16 holds k0+16..k0+31. Base already includes +16*half.
__device__ __forceinline__ bf16x16 load_fragA(const char* base) {
  FragU u;
  u.q[0] = *(const u32x4*)(base);
  u.q[1] = *(const u32x4*)(base + 16);
  return u.v;
}
// B-matrix 32x16 bf16, stored as B^T rows (N-major, contiguous K): per lane
// N = lane%16, K = k0+16*half .. +15.
__device__ __forceinline__ bf16x16 load_fragB(const char* base) {
  FragU u;
  u.q[0] = *(const u32x4*)(base);
  u.q[1] = *(const u32x4*)(base + 16);
  return u.v;
}

// ---------------------------------------------------------------------------
// One-time (per launch) weight conversion: fp32 row-major -> bf16 transposed.
//   Wv,Wg: [E][INNER] -> WvT,WgT: [INNER][E]   (B^T for stage-1 GEMMs)
//   Wo:    [INNER][E] -> WoT:     [E][INNER]   (B^T for stage-2 GEMM)
// 192 KB total -> permanently resident in the 192 MB L2.
// ---------------------------------------------------------------------------
__global__ __launch_bounds__(256)
void wconvert_kernel(const float* __restrict__ Wv, const float* __restrict__ Wg,
                     const float* __restrict__ Wo,
                     __bf16* __restrict__ WvT, __bf16* __restrict__ WgT,
                     __bf16* __restrict__ WoT) {
  const int t = blockIdx.x * 256 + threadIdx.x;   // 0..32767
  const int n  = t >> 7, k  = t & 127;            // [INNER][E]
  WvT[n * EDIM + k] = (__bf16)Wv[k * INNERD + n];
  WgT[n * EDIM + k] = (__bf16)Wg[k * INNERD + n];
  const int n2 = t >> 8, k2 = t & 255;            // [E][INNER]
  WoT[n2 * INNERD + k2] = (__bf16)Wo[k2 * EDIM + n2];
}

// ---------------------------------------------------------------------------
// Fused main kernel: 256 threads = 8 waves, 64 IE-rows per block.
// ---------------------------------------------------------------------------
__global__ __launch_bounds__(256)
void ie_atten_fused_kernel(const float* __restrict__ IE,
                           const __bf16* __restrict__ WvT,
                           const __bf16* __restrict__ WgT,
                           const __bf16* __restrict__ WoT,
                           const float* __restrict__ bg,
                           const float* __restrict__ bo,
                           float* __restrict__ out) {
  __shared__ __align__(16) __bf16 Xb[MT * EDIM];    // 16 KB bf16 X tile
  __shared__ __align__(16) __bf16 Gs[MT * INNERD];  // 32 KB bf16 sigmoid(gate)
  __shared__ __align__(16) __bf16 Mb[MT * INNERD];  // 32 KB bf16 gated V

  const int tid  = threadIdx.x;
  const int wid  = tid >> 5;
  const int lane = tid & 31;
  const int ln   = lane & 15;
  const int hf   = lane >> 4;
  const long r0  = (long)blockIdx.x * MT;

  // ---- stage 0: stage 64x128 fp32 rows of IE into LDS as bf16 -------------
  {
#pragma unroll
    for (int it = 0; it < 2; ++it) {
      const int u = tid + it * 256;        // 512 row-segments of 16 floats
      const int row = u >> 3, seg = u & 7;
      const f32x4* src = (const f32x4*)(IE + (r0 + row) * EDIM + seg * 16);
      __bf16* dst = Xb + row * EDIM + seg * 16;
#pragma unroll
      for (int q = 0; q < 4; ++q) {
        f32x4 f = src[q];
        dst[q * 4 + 0] = (__bf16)f[0];
        dst[q * 4 + 1] = (__bf16)f[1];
        dst[q * 4 + 2] = (__bf16)f[2];
        dst[q * 4 + 3] = (__bf16)f[3];
      }
    }
  }
  __syncthreads();

  // ---- stage 1a: Gs = sigmoid(X @ Wg + bg), bf16 in LDS -------------------
  // wave wid owns ntiles {2*wid, 2*wid+1} of INNER=256; mtiles 0..3.
  {
    f32x8 acc[2][NMT] = {};
#pragma unroll
    for (int kc = 0; kc < 4; ++kc) {
      const int k0 = kc * 32;
      bf16x16 a[NMT];
#pragma unroll
      for (int m = 0; m < NMT; ++m)
        a[m] = load_fragA((const char*)Xb + ((m * 16 + ln) * EDIM + k0 + 16 * hf) * 2);
#pragma unroll
      for (int n = 0; n < 2; ++n) {
        const int nt = wid * 2 + n;
        bf16x16 b = load_fragB((const char*)WgT + ((nt * 16 + ln) * EDIM + k0 + 16 * hf) * 2);
#pragma unroll
        for (int m = 0; m < NMT; ++m)
          acc[n][m] = __builtin_amdgcn_wmma_f32_16x16x32_bf16(
              false, a[m], false, b, (short)0, acc[n][m], false, false);
      }
    }
#pragma unroll
    for (int n = 0; n < 2; ++n) {
      const int c = (wid * 2 + n) * 16 + ln;   // D layout: N = lane%16
      const float bgv = bg[c];
#pragma unroll
      for (int m = 0; m < NMT; ++m)
#pragma unroll
        for (int r = 0; r < 8; ++r) {
          const int row = m * 16 + r + 8 * hf; // D layout: M = r + 8*(lane/16)
          const float x = acc[n][m][r] + bgv;
          Gs[row * INNERD + c] = (__bf16)(1.0f / (1.0f + __expf(-x)));
        }
    }
  }
  __syncthreads();

  // ---- stage 1b: Mb = bf16( (X @ Wv) * Gs[gate-perm] ) --------------------
  {
    f32x8 acc[2][NMT] = {};
#pragma unroll
    for (int kc = 0; kc < 4; ++kc) {
      const int k0 = kc * 32;
      bf16x16 a[NMT];
#pragma unroll
      for (int m = 0; m < NMT; ++m)
        a[m] = load_fragA((const char*)Xb + ((m * 16 + ln) * EDIM + k0 + 16 * hf) * 2);
#pragma unroll
      for (int n = 0; n < 2; ++n) {
        const int nt = wid * 2 + n;
        bf16x16 b = load_fragB((const char*)WvT + ((nt * 16 + ln) * EDIM + k0 + 16 * hf) * 2);
#pragma unroll
        for (int m = 0; m < NMT; ++m)
          acc[n][m] = __builtin_amdgcn_wmma_f32_16x16x32_bf16(
              false, a[m], false, b, (short)0, acc[n][m], false, false);
      }
    }
#pragma unroll
    for (int n = 0; n < 2; ++n) {
      const int c = (wid * 2 + n) * 16 + ln;       // c = h*64 + d
      const int gcol = ((c & 63) * 4) + (c >> 6);  // gate stored as (d h)
#pragma unroll
      for (int m = 0; m < NMT; ++m)
#pragma unroll
        for (int r = 0; r < 8; ++r) {
          const int row = m * 16 + r + 8 * hf;
          const float mv = acc[n][m][r] * (float)Gs[row * INNERD + gcol];
          Mb[row * INNERD + c] = (__bf16)mv;
        }
    }
  }
  __syncthreads();

  // ---- stage 2: out = Mb @ Wo + bo ----------------------------------------
  // wave wid owns ntile wid of E=128; K = INNER = 256 (8 chunks), mtiles 0..3.
  {
    f32x8 acc[NMT] = {};
    const int nt = wid;
#pragma unroll
    for (int kc = 0; kc < 8; ++kc) {
      const int k0 = kc * 32;
      bf16x16 b = load_fragB((const char*)WoT + ((nt * 16 + ln) * INNERD + k0 + 16 * hf) * 2);
#pragma unroll
      for (int m = 0; m < NMT; ++m) {
        bf16x16 a = load_fragA((const char*)Mb + ((m * 16 + ln) * INNERD + k0 + 16 * hf) * 2);
        acc[m] = __builtin_amdgcn_wmma_f32_16x16x32_bf16(
            false, a, false, b, (short)0, acc[m], false, false);
      }
    }
    const int c = nt * 16 + ln;
    const float bov = bo[c];
#pragma unroll
    for (int m = 0; m < NMT; ++m)
#pragma unroll
      for (int r = 0; r < 8; ++r) {
        const int row = m * 16 + r + 8 * hf;
        out[(r0 + row) * EDIM + c] = acc[m][r] + bov;
      }
  }
}

// ---------------------------------------------------------------------------
// Inputs (setup_inputs order): 0 inter_edges, 1 ab_mask, 2 at_mask, 3 Wq,
// 4 Wk, 5 Wv, 6 Web, 7 Wg, 8 bg, 9 Wo, 10 bo.
// Masks / Wq / Wk / Web are mathematically dead (softmax row-sum == 1).
// ---------------------------------------------------------------------------
extern "C" void kernel_launch(void* const* d_in, const int* in_sizes, int n_in,
                              void* d_out, int out_size, void* d_ws, size_t ws_size,
                              hipStream_t stream) {
  const float* IE = (const float*)d_in[0];
  const float* Wv = (const float*)d_in[5];
  const float* Wg = (const float*)d_in[7];
  const float* bg = (const float*)d_in[8];
  const float* Wo = (const float*)d_in[9];
  const float* bo = (const float*)d_in[10];

  __bf16* WvT = (__bf16*)d_ws;                 //  64 KB
  __bf16* WgT = WvT + INNERD * EDIM;           //  64 KB
  __bf16* WoT = WgT + INNERD * EDIM;           //  64 KB

  const long nrows = (long)in_sizes[0] / EDIM; // b*i*j = 65536
  const int  nblocks = (int)(nrows / MT);      // 1024

  wconvert_kernel<<<(INNERD * EDIM) / 256, 256, 0, stream>>>(Wv, Wg, Wo, WvT, WgT, WoT);
  ie_atten_fused_kernel<<<nblocks, 256, 0, stream>>>(IE, WvT, WgT, WoT, bg, bo,
                                                     (float*)d_out);
}
